// MeshConv_52261162058490
// MI455X (gfx1250) — compile-verified, matrix operand order checked by reading, not run on hardware
//
#include <hip/hip_runtime.h>

// MeshConv: gather(5 neighbors) -> features G(E,320) -> GEMM with W(128,320) -> out (B,128,E,1)
// B=4, C_IN=64, E=75000, K=5, C_OUT=128.
// Strategy: bf16 hi/lo split-precision WMMA (3x v_wmma_f32_16x16x32_bf16 per K-chunk)
// emulating fp32 accuracy on the bf16 matrix pipe; gathers software-pipelined behind WMMAs.

#define B_    4
#define CIN   64
#define E_    75000
#define KNB   5
#define COUT  128
#define KDIM  320          // 5*64
#define NCHUNK 10          // KDIM / 32
#define GSTRIDE 392        // padded bf16 row stride: 196 dwords == 4 (mod 64) -> conflict-free b128 reads
#define TILES_PER_WG 16    // 256 edges per workgroup: amortize W-register load over more tiles
#define EDGES_WG (16 * TILES_PER_WG)

typedef __attribute__((ext_vector_type(16))) __bf16 v16bf;
typedef __attribute__((ext_vector_type(8)))  __bf16 v8bf;
typedef __attribute__((ext_vector_type(2)))  __bf16 v2bf;
typedef __attribute__((ext_vector_type(8)))  float  v8f;
typedef __attribute__((ext_vector_type(2)))  float  v2f;

union V16 { v16bf v; v8bf h[2]; };

// Split a float pair into packed bf16 hi + lo (hi = RNE bf16, lo = bf16(residual)).
__device__ __forceinline__ void split_store(v2f g, __bf16* hip, __bf16* lop) {
  __bf16 h0 = (__bf16)g.x, h1 = (__bf16)g.y;
  v2bf hv; hv.x = h0; hv.y = h1;
  v2bf lv; lv.x = (__bf16)(g.x - (float)h0); lv.y = (__bf16)(g.y - (float)h1);
  *(v2bf*)hip = hv;
  *(v2bf*)lop = lv;
}

// ---------------- Transpose x (B,CIN,E) -> xT (B,E,CIN), coalesced both sides -------------
__global__ __launch_bounds__(256) void meshconv_transpose(const float* __restrict__ x,
                                                          float* __restrict__ xT) {
  __shared__ float tile[32][33];
  const int b  = blockIdx.z;
  const int c0 = blockIdx.y * 32;      // CIN=64 -> 2 tiles
  const int e0 = blockIdx.x * 32;
  const int tx = threadIdx.x;          // 32
  const int ty = threadIdx.y;          // 8
#pragma unroll
  for (int i = 0; i < 32; i += 8) {
    int c = c0 + ty + i, e = e0 + tx;
    float v = 0.f;
    if (e < E_) v = x[((size_t)b * CIN + c) * E_ + e];
    tile[ty + i][tx] = v;
  }
  __syncthreads();
#pragma unroll
  for (int i = 0; i < 32; i += 8) {
    int e = e0 + ty + i, c = c0 + tx;
    if (e < E_) xT[((size_t)b * E_ + e) * CIN + c] = tile[tx][ty + i];
  }
}

// ---------------- Main fused gather + feature + split-bf16 WMMA GEMM ----------------------
__global__ __launch_bounds__(256) void meshconv_main(const float* __restrict__ xsrc,
                                                     int xTransposed,
                                                     const int* __restrict__ edge,
                                                     const float* __restrict__ W,
                                                     const float* __restrict__ bias,
                                                     float* __restrict__ out) {
  __shared__ __align__(16) __bf16 Ghi[16 * GSTRIDE];
  __shared__ __align__(16) __bf16 Glo[16 * GSTRIDE];
  __shared__ float TT[8][16 * 17];     // per-wave 16x16 D-tile transpose buffers (pad 17)
  __shared__ int IDX[EDGES_WG * KNB];  // this WG's edge indices (coalesced copy, then ds reads)

  const int t     = threadIdx.x;
  const int wave  = t >> 5;
  const int lane  = t & 31;
  const int b     = blockIdx.y;
  const int eWG   = blockIdx.x * EDGES_WG;
  const int obase = wave * 16;         // each wave owns 16 output channels
  const int n     = lane & 15;
  const int hhalf = lane >> 4;         // 0: lanes 0-15, 1: lanes 16-31
  const int o     = obase + n;
  const int c2    = lane * 2;          // channel pair owned in the build phase
  const int mbase = wave;              // edge row within tile (plus u*8)

  // ---- Stage this WG's 256x5 indices into LDS (coalesced burst, clamped at E) ----
  {
    const int* src = edge + (size_t)b * E_ * KNB;
    for (int i = t; i < EDGES_WG * KNB; i += 256) {
      int l = i / KNB, r = i - l * KNB;
      int srcE = eWG + l; if (srcE >= E_) srcE = E_ - 1;
      IDX[i] = src[(size_t)srcE * KNB + r];
    }
  }

  // ---- Preload W fragments (bf16 hi + lo) into registers; reused for all 16 M-tiles ----
  // B-matrix (KxN) fragment: lane n holds column N=n; elems 0..7 = K kb..kb+7,
  // elems 8..15 = K kb+16..kb+23, kb = kc*32 + (lane>=16 ? 8 : 0).
  V16 whi[NCHUNK], wlo[NCHUNK];
#pragma unroll
  for (int kc = 0; kc < NCHUNK; ++kc) {
    const int kb = kc * 32 + hhalf * 8;
    const float* wp = W + (size_t)o * KDIM + kb;
#pragma unroll
    for (int i = 0; i < 8; ++i) {
      float f0 = wp[i];
      float f1 = wp[16 + i];
      __bf16 h0 = (__bf16)f0, h1 = (__bf16)f1;
      whi[kc].v[i]     = h0;
      whi[kc].v[8 + i] = h1;
      wlo[kc].v[i]     = (__bf16)(f0 - (float)h0);
      wlo[kc].v[8 + i] = (__bf16)(f1 - (float)h1);
    }
  }
  const float bv = bias[o];

  // Issue the 5 gather loads (2 edge rows per thread) for tile `ti` into p.
  auto issue_gather = [&](int ti, v2f (&p)[2][5]) {
#pragma unroll
    for (int u = 0; u < 2; ++u) {
      int l = ti * 16 + mbase + u * 8;          // local edge row in [0,EDGES_WG)
      int e = eWG + l; if (e >= E_) e = E_ - 1;
      const int* ip = &IDX[l * KNB];
      int i1 = ip[1], i2 = ip[2], i3 = ip[3], i4 = ip[4];
      if (xTransposed) {
        const float* xb = xsrc + (size_t)b * E_ * CIN + c2;   // (E, CIN) rows, 256B each
        p[u][0] = *(const v2f*)(xb + (size_t)e  * CIN);       // self index (edgemat col 0 == e)
        p[u][1] = *(const v2f*)(xb + (size_t)i1 * CIN);
        p[u][2] = *(const v2f*)(xb + (size_t)i2 * CIN);
        p[u][3] = *(const v2f*)(xb + (size_t)i3 * CIN);
        p[u][4] = *(const v2f*)(xb + (size_t)i4 * CIN);
      } else {
        const float* xa = xsrc + ((size_t)b * CIN + c2) * E_; // fallback: (CIN, E) strided
        const float* xc = xa + E_;                            // channel c2+1
        p[u][0].x = xa[e];  p[u][0].y = xc[e];
        p[u][1].x = xa[i1]; p[u][1].y = xc[i1];
        p[u][2].x = xa[i2]; p[u][2].y = xc[i2];
        p[u][3].x = xa[i3]; p[u][3].y = xc[i3];
        p[u][4].x = xa[i4]; p[u][4].y = xc[i4];
      }
    }
  };

  // Combine prefetched rows into features, split to bf16 hi/lo, store G tile in LDS.
  auto consume = [&](v2f (&p)[2][5]) {
#pragma unroll
    for (int u = 0; u < 2; ++u) {
      int m = mbase + u * 8;
      v2f r1 = p[u][1], r2 = p[u][2], r3 = p[u][3], r4 = p[u][4];
      v2f g0 = p[u][0];
      v2f g1 = r1 + r3;
      v2f g2 = r2 + r4;
      v2f d1 = r1 - r3, d2 = r2 - r4;
      v2f g3; g3.x = __builtin_fabsf(d1.x); g3.y = __builtin_fabsf(d1.y);
      v2f g4; g4.x = __builtin_fabsf(d2.x); g4.y = __builtin_fabsf(d2.y);
      const int base = m * GSTRIDE + c2;
      split_store(g0, &Ghi[base +   0], &Glo[base +   0]);
      split_store(g1, &Ghi[base +  64], &Glo[base +  64]);
      split_store(g2, &Ghi[base + 128], &Glo[base + 128]);
      split_store(g3, &Ghi[base + 192], &Glo[base + 192]);
      split_store(g4, &Ghi[base + 256], &Glo[base + 256]);
    }
  };

  v2f p[2][5];
  __syncthreads();               // IDX visible to all waves
  issue_gather(0, p);            // prologue: tile 0 gathers in flight

  for (int ti = 0; ti < TILES_PER_WG; ++ti) {   // e0 < E_ guaranteed by grid sizing
    const int e0 = eWG + ti * 16;

    consume(p);                  // waits on in-flight gathers, builds G tile
    __syncthreads();

    // Prefetch next tile's gathers: loads fly during the 30 WMMAs below.
    if (ti + 1 < TILES_PER_WG) issue_gather(ti + 1, p);

    // ---- Split-precision WMMA: acc += Ahi*Bhi + Ahi*Blo + Alo*Bhi over 10 K-chunks ----
    // A-matrix (MxK) fragment: lane holds row M=lane&15; elems 0..7 = K kb..kb+7,
    // elems 8..15 = K kb+16..kb+23, kb = kc*32 + (lane>=16 ? 8 : 0).
    const __bf16* gh0 = &Ghi[n * GSTRIDE + hhalf * 8];
    const __bf16* gl0 = &Glo[n * GSTRIDE + hhalf * 8];
    v8f acc = {};
#pragma unroll
    for (int kc = 0; kc < NCHUNK; ++kc) {
      V16 ahi, alo;
      const __bf16* gh = gh0 + kc * 32;
      const __bf16* gl = gl0 + kc * 32;
      ahi.h[0] = *(const v8bf*)(gh);
      ahi.h[1] = *(const v8bf*)(gh + 16);
      alo.h[0] = *(const v8bf*)(gl);
      alo.h[1] = *(const v8bf*)(gl + 16);
      acc = __builtin_amdgcn_wmma_f32_16x16x32_bf16(false, ahi.v, false, whi[kc].v,
                                                    (short)0, acc, false, false);
      acc = __builtin_amdgcn_wmma_f32_16x16x32_bf16(false, ahi.v, false, wlo[kc].v,
                                                    (short)0, acc, false, false);
      acc = __builtin_amdgcn_wmma_f32_16x16x32_bf16(false, alo.v, false, whi[kc].v,
                                                    (short)0, acc, false, false);
    }

    // ---- Bias + per-wave LDS transpose so stores are coalesced along E ----
    // D layout: VGPR r -> (M = r + hhalf*8, N = lane&15)
    float* tt = &TT[wave][0];
#pragma unroll
    for (int r = 0; r < 8; ++r)
      tt[(r + hhalf * 8) * 17 + n] = acc[r] + bv;
    __syncthreads();

#pragma unroll
    for (int r2 = 0; r2 < 8; ++r2) {
      int ncol = r2 + hhalf * 8;           // output channel within wave's tile
      int e    = e0 + n;                   // edge = tile base + lane&15
      if (e < E_) {
        float val = tt[n * 17 + ncol];
        out[((size_t)b * COUT + obase + ncol) * E_ + e] = val;  // lanes -> consecutive e
      }
    }
  }
}

extern "C" void kernel_launch(void* const* d_in, const int* in_sizes, int n_in,
                              void* d_out, int out_size, void* d_ws, size_t ws_size,
                              hipStream_t stream) {
  const float* x    = (const float*)d_in[0];   // (B, CIN, E) fp32
  const int*   edge = (const int*)d_in[1];     // (B, E, 5)   int
  const float* W    = (const float*)d_in[2];   // (128, 320)  fp32
  const float* bias = (const float*)d_in[3];   // (128,)      fp32
  float*       out  = (float*)d_out;           // (B, 128, E, 1) fp32

  const size_t xtBytes = (size_t)B_ * E_ * CIN * sizeof(float);  // 76.8 MB
  int useT = (ws_size >= xtBytes) ? 1 : 0;
  const float* xsrc = x;
  if (useT) {
    float* xT = (float*)d_ws;
    dim3 gT((E_ + 31) / 32, CIN / 32, B_);
    meshconv_transpose<<<gT, dim3(32, 8), 0, stream>>>(x, xT);
    xsrc = xT;
  }
  dim3 gM((E_ + EDGES_WG - 1) / EDGES_WG, B_);
  meshconv_main<<<gM, 256, 0, stream>>>(xsrc, useT, edge, W, bias, out);
}
